// Recommender_6176162972142
// MI455X (gfx1250) — matrix operand order repaired and checked by reading, not compiled, not run on hardware
//
#include <hip/hip_runtime.h>

typedef __attribute__((ext_vector_type(16))) __bf16 v16bf;
typedef __attribute__((ext_vector_type(8)))  float  v8f;
typedef unsigned short ushort_t;
typedef unsigned int   uint_t;

#define DIMV 64
#define BB   32
#define SS   20
#define LL   50
#define NN   40000
#define NT   64        // candidates per block in stage B
#define AST  68        // padded LDS row stride (ushorts) to avoid bank conflicts

union Frag { uint_t u[8]; v16bf v; };

__device__ __forceinline__ ushort_t f2bf(float f) {
    uint_t x = __float_as_uint(f);
    uint_t r = (x + 0x7FFFu + ((x >> 16) & 1u)) >> 16;   // round-to-nearest-even
    return (ushort_t)r;
}

// ---------------- Stage A: per-batch small work ----------------
__global__ __launch_bounds__(64) void recsys_stageA(
    const int* __restrict__ user_ids, const int* __restrict__ sess_item,
    const int* __restrict__ rcnt_item,
    const float* __restrict__ item_emb, const float* __restrict__ user_emb,
    const float* __restrict__ user_emb2,
    const float* __restrict__ ue_w1, const float* __restrict__ ue_b1,
    const float* __restrict__ ue_w2, const float* __restrict__ ue_b2,
    const float* __restrict__ uenc_w, const float* __restrict__ uenc_b,
    const float* __restrict__ pred_w, const float* __restrict__ pred_b,
    float* __restrict__ out_hu, float* __restrict__ out_hc,
    ushort_t* __restrict__ wsA, float* __restrict__ wsC1)
{
    __shared__ float uemb[DIMV], u2[DIMV], hs[LL][DIMV], wl[LL], part[DIMV];
    __shared__ float ee[SS][DIMV], att[SS][SS], lg[SS][SS];
    __shared__ float hu[DIMV], hr[DIMV], hc[SS][DIMV];

    const int b = blockIdx.x, d = threadIdx.x;
    const int uid = user_ids[b];
    uemb[d] = user_emb[uid * DIMV + d];
    u2[d]   = user_emb2[uid * DIMV + d];
    for (int l = 0; l < LL; ++l)
        hs[l][d] = item_emb[rcnt_item[b * LL + l] * DIMV + d];
    __syncthreads();

    // w = relu([uemb,hs] @ ue_w1 + b1) @ ue_w2 + b2
    for (int l = 0; l < LL; ++l) {
        float acc = ue_b1[d];
        for (int k = 0; k < DIMV; ++k) acc += uemb[k]    * ue_w1[k * DIMV + d];
        for (int k = 0; k < DIMV; ++k) acc += hs[l][k]   * ue_w1[(DIMV + k) * DIMV + d];
        part[d] = fmaxf(acc, 0.f) * ue_w2[d];
        __syncthreads();
        if (d == 0) { float s = ue_b2[0]; for (int k = 0; k < DIMV; ++k) s += part[k]; wl[l] = s; }
        __syncthreads();
    }
    if (d == 0) {   // masked stable softmax over L
        float m = -1e30f;
        for (int l = 0; l < LL; ++l) m = fmaxf(m, wl[l]);
        float ssum = 0.f;
        for (int l = 0; l < LL; ++l) {
            float ev = expf(wl[l] - m) + 1e-10f;
            if (rcnt_item[b * LL + l] == 0) ev = 0.f;
            wl[l] = ev; ssum += ev;
        }
        for (int l = 0; l < LL; ++l) wl[l] /= ssum;
    }
    __syncthreads();
    { float acc = 0.f; for (int l = 0; l < LL; ++l) acc += wl[l] * hs[l][d]; hr[d] = acc; }
    __syncthreads();
    { float acc = uenc_b[d];
      for (int k = 0; k < DIMV; ++k) acc += u2[k] * uenc_w[k * DIMV + d];
      for (int k = 0; k < DIMV; ++k) acc += hr[k] * uenc_w[(DIMV + k) * DIMV + d];
      hu[d] = acc; out_hu[b * DIMV + d] = acc; }

    // session self-attention
    for (int s = 0; s < SS; ++s) ee[s][d] = item_emb[sess_item[b * SS + s] * DIMV + d];
    __syncthreads();
    for (int idx = d; idx < SS * SS; idx += 64) {
        int si = idx / SS, ti = idx % SS;
        float acc = 0.f;
        for (int k = 0; k < DIMV; ++k) acc += ee[si][k] * ee[ti][k];
        lg[si][ti] = acc * 0.125f;     // 1/sqrt(64)
    }
    __syncthreads();
    if (d < SS) {
        int si = d;
        bool ms = sess_item[b * SS + si] != 0;
        float m = -1e30f;
        for (int t = 0; t < SS; ++t) m = fmaxf(m, lg[si][t]);
        float ssum = 0.f;
        for (int t = 0; t < SS; ++t) {
            float ev = expf(lg[si][t] - m) + 1e-10f;
            bool mt = sess_item[b * SS + t] != 0;
            if (!(ms && mt)) ev = 0.f;
            att[si][t] = ev; ssum += ev;
        }
        for (int t = 0; t < SS; ++t) att[si][t] /= ssum;
    }
    __syncthreads();
    for (int s = 0; s < SS; ++s) {
        float acc = 0.f;
        for (int t = 0; t < SS; ++t) acc += att[s][t] * ee[t][d];
        hc[s][d] = acc; out_hc[(b * SS + s) * DIMV + d] = acc;
    }
    __syncthreads();
    if (d < SS) {  // c1[b,s] = hu.wu + hc_s.wc + pred_b
        float acc = pred_b[0];
        for (int k = 0; k < DIMV; ++k) acc += hu[k]    * pred_w[k];
        for (int k = 0; k < DIMV; ++k) acc += hc[d][k] * pred_w[DIMV + k];
        wsC1[b * SS + d] = acc;
    }
    // pack A (32x64 bf16): rows 0..19 = hc, 20 = hu, 21 = wv, 22..31 = 0
    for (int x = d; x < 32 * DIMV; x += 64) {
        int r = x >> 6, k = x & 63;
        float v = (r < SS) ? hc[r][k] : (r == SS) ? hu[k]
                 : (r == SS + 1) ? pred_w[2 * DIMV + k] : 0.f;
        wsA[b * 32 * DIMV + x] = f2bf(v);
    }
}

// ---------------- Stage B: WMMA over candidates ----------------
__global__ __launch_bounds__(128) void recsys_stageB(
    const int* __restrict__ items, const float* __restrict__ item_emb,
    const ushort_t* __restrict__ wsA, const float* __restrict__ wsC1,
    float* __restrict__ out_scores, float* __restrict__ out_attn)
{
    __shared__ alignas(16) ushort_t A_sh[32 * AST];
    __shared__ alignas(16) ushort_t V_sh[NT * AST];
    __shared__ int   ids[NT];
    __shared__ float c1[SS];

    const int tid = threadIdx.x;
    const int nTiles = NN / NT;                 // 625
    const int b  = blockIdx.x / nTiles;
    const int nb = (blockIdx.x % nTiles) * NT;

    if (tid < NT) ids[tid] = items[b * NN + nb + tid];
    if (tid < SS) c1[tid]  = wsC1[b * SS + tid];
    for (int x = tid; x < 32 * DIMV; x += 128)
        A_sh[(x >> 6) * AST + (x & 63)] = wsA[b * 32 * DIMV + x];
    __syncthreads();
    for (int x = tid; x < NT * DIMV; x += 128) {
        int i = x >> 6, k = x & 63;
        V_sh[i * AST + k] = f2bf(item_emb[ids[i] * DIMV + k]);
    }
    __syncthreads();

    const int wv   = tid >> 5;         // wave id 0..3 -> 16-col sub-tile
    const int lane = tid & 31;
    const int h    = lane >> 4;
    const int m0   = lane & 15;        // A-row / B-col within tile

    Frag a0[2], a1[2], bf[2];
    #pragma unroll
    for (int c = 0; c < 2; ++c) {
        #pragma unroll
        for (int j = 0; j < 8; ++j) {
            int ka = 32 * c + 16 * h + 2 * j;
            a0[c].u[j] = *(const uint_t*)&A_sh[m0 * AST + ka];
            a1[c].u[j] = *(const uint_t*)&A_sh[(16 + m0) * AST + ka];
            int kb = 32 * c + 16 * h + 2 * j;
            bf[c].u[j] = *(const uint_t*)&V_sh[(wv * 16 + m0) * AST + kb];
        }
    }

    v8f acc0 = {}, acc1 = {};
    acc0 = __builtin_amdgcn_wmma_f32_16x16x32_bf16(false, a0[0].v, false, bf[0].v, (short)0, acc0, false, false);
    acc0 = __builtin_amdgcn_wmma_f32_16x16x32_bf16(false, a0[1].v, false, bf[1].v, (short)0, acc0, false, false);
    acc1 = __builtin_amdgcn_wmma_f32_16x16x32_bf16(false, a1[0].v, false, bf[0].v, (short)0, acc1, false, false);
    acc1 = __builtin_amdgcn_wmma_f32_16x16x32_bf16(false, a1[1].v, false, bf[1].v, (short)0, acc1, false, false);

    // rows 20 (hu_v) / 21 (vemb.wv) live in acc1[4]/acc1[5] of the h==0 half
    float huv = __shfl(acc1[4], m0, 32);
    float wvd = __shfl(acc1[5], m0, 32);

    const int n = nb + wv * 16 + m0;
    #pragma unroll
    for (int r = 0; r < 8; ++r) {      // s = 0..15
        int s0 = r + 8 * h;
        float g  = c1[s0] + wvd;
        float at = 1.f / (1.f + expf(-g));
        float sc = at * huv + (1.f - at) * acc0[r];
        size_t off = ((size_t)(b * SS + s0)) * NN + n;
        out_scores[off] = sc;
        out_attn[off]   = at;
    }
    if (h == 0) {
        #pragma unroll
        for (int r = 0; r < 4; ++r) {  // s = 16..19
            int s1 = 16 + r;
            float g  = c1[s1] + wvd;
            float at = 1.f / (1.f + expf(-g));
            float sc = at * huv + (1.f - at) * acc1[r];
            size_t off = ((size_t)(b * SS + s1)) * NN + n;
            out_scores[off] = sc;
            out_attn[off]   = at;
        }
    }
}

extern "C" void kernel_launch(void* const* d_in, const int* in_sizes, int n_in,
                              void* d_out, int out_size, void* d_ws, size_t ws_size,
                              hipStream_t stream) {
    const int*   user_ids  = (const int*)  d_in[0];
    const int*   sess_item = (const int*)  d_in[1];
    const int*   rcnt_item = (const int*)  d_in[2];
    const int*   items     = (const int*)  d_in[3];
    const float* item_emb  = (const float*)d_in[4];
    const float* user_emb  = (const float*)d_in[5];
    const float* user_emb2 = (const float*)d_in[6];
    const float* ue_w1     = (const float*)d_in[7];
    const float* ue_b1     = (const float*)d_in[8];
    const float* ue_w2     = (const float*)d_in[9];
    const float* ue_b2     = (const float*)d_in[10];
    const float* uenc_w    = (const float*)d_in[11];
    const float* uenc_b    = (const float*)d_in[12];
    const float* pred_w    = (const float*)d_in[13];
    const float* pred_b    = (const float*)d_in[14];

    float* out = (float*)d_out;
    const size_t scores_off = 0;
    const size_t hu_off     = (size_t)BB * SS * NN;            // 25,600,000
    const size_t hc_off     = hu_off + (size_t)BB * DIMV;      // +2048
    const size_t attn_off   = hc_off + (size_t)BB * SS * DIMV; // +40,960

    ushort_t* wsA  = (ushort_t*)d_ws;                              // 32*32*64 bf16
    float*    wsC1 = (float*)((char*)d_ws + (size_t)BB * 32 * DIMV * sizeof(ushort_t));

    recsys_stageA<<<BB, 64, 0, stream>>>(
        user_ids, sess_item, rcnt_item, item_emb, user_emb, user_emb2,
        ue_w1, ue_b1, ue_w2, ue_b2, uenc_w, uenc_b, pred_w, pred_b,
        out + hu_off, out + hc_off, wsA, wsC1);

    recsys_stageB<<<BB * (NN / NT), 128, 0, stream>>>(
        items, item_emb, wsA, wsC1,
        out + scores_off, out + attn_off);
}
